// DynamicMatrixRouter_38371237822637
// MI455X (gfx1250) — matrix operand order repaired and checked by reading, not compiled, not run on hardware
//
#include <hip/hip_runtime.h>
#include <hip/hip_bf16.h>
#include <cstdint>

// ---------------------------------------------------------------------------
// DynamicMatrixRouter for MI455X (gfx1250, wave32, WMMA + TDM)
//   N = B*S = 16384 tokens, D = 4096, H = 2048, E = 64 experts, K = 3
// Dominant GEMM (x@W1, 275 GFLOP) runs as bf16 V_WMMA_F32_16X16X32_BF16 with
// f32 accumulation; relu + dot(W2) fused into the epilogue so h[N,H] is never
// materialized. Gate GEMM (x@W) also WMMA. The 16-token x tile is staged into
// LDS (fp32) by the Tensor Data Mover (tensor_load_to_lds + s_wait_tensorcnt),
// and converted to bf16 fragments on the fly with v_cvt_pk_bf16_f32.
// ---------------------------------------------------------------------------

typedef __attribute__((ext_vector_type(16))) __bf16        v16bf;
typedef __attribute__((ext_vector_type(8)))  float         v8f;
typedef __attribute__((ext_vector_type(4)))  unsigned int  u32x4;
typedef __attribute__((ext_vector_type(8)))  int           i32x8;
typedef __attribute__((ext_vector_type(4)))  int           i32x4;

namespace dmr {
constexpr int D    = 4096;   // model dim
constexpr int H    = 2048;   // predictor hidden
constexpr int E    = 64;     // experts
constexpr int N    = 16384;  // tokens (B*S)
constexpr int TOPK = 3;
// TDM pads the LDS destination: 4 DWORDs after every 256 DWORDs
// (pad_interval code 7, pad_amount code 3) -> row stride 4160 DWORDs,
// and every 32-element fragment stays 16-byte aligned.
constexpr int PAD_CHUNK = 256;                        // DWORDs between pads
constexpr int PAD_DW    = 4;                          // pad DWORDs
constexpr int ROWDW     = D + (D / PAD_CHUNK) * PAD_DW; // 4160
constexpr unsigned XTILE_BYTES = 16u * ROWDW * 4u;      // 266,240 B
__device__ __forceinline__ int pdw(int k) {             // padded dword index
  return k + ((k >> 8) << 2);
}
}

#if defined(__has_builtin)
#if __has_builtin(__builtin_amdgcn_tensor_load_to_lds)
#define DMR_HAS_TDM 1
#else
#define DMR_HAS_TDM 0
#endif
#else
#define DMR_HAS_TDM 0
#endif

#if __has_include(<hip/amd_detail/amd_gfx1250_TDM.h>)
#define DMR_TDM_6ARG 1   // therock-10.0 headers -> clang-23 6-arg builtin
#else
#define DMR_TDM_6ARG 0   // ROCm 7.2 -> clang-22 5-arg builtin
#endif

// Stage one 16-token x tile (fp32) into LDS offset 0 via the Tensor Data
// Mover. Only wave 0 issues the DMA; everyone syncs on the barrier.
__device__ __forceinline__ void stage_x_tile(const float* __restrict__ x,
                                             long tok0, float* xs,
                                             int wave, int tid) {
  using namespace dmr;
#if DMR_HAS_TDM
  if (wave == 0) {
    const unsigned long long ga =
        (unsigned long long)(uintptr_t)(x + tok0 * (long)D);
    u32x4 g0;
    g0[0] = 1u;                                   // count=1, user descriptor
    g0[1] = 0u;                                   // lds_addr (byte offset)
    g0[2] = (unsigned)(ga & 0xffffffffull);       // global_addr[31:0]
    g0[3] = (unsigned)((ga >> 32) & 0x1ffffffull) // global_addr[56:32]
            | (2u << 30);                         // type = 2 ("image")
    i32x8 g1;
    g1[0] = (int)((2u << 16)    // data_size = 4 bytes
                | (1u << 20)    // pad_enable
                | (7u << 22)    // pad_interval: 256 DWORDs
                | (3u << 25));  // pad_amount: 4 DWORDs
    g1[1] = (int)((unsigned)D << 16);     // tensor_dim0[15:0]  @ bits[63:48]
    g1[2] = (int)((unsigned)N << 16);     // tensor_dim1[15:0]  @ bits[95:80]
    g1[3] = (int)((unsigned)D << 16);     // tile_dim0          @ bits[127:112]
    g1[4] = 16;                           // tile_dim1 = 16 rows
    g1[5] = D;                            // tensor_dim0_stride low32
    g1[6] = 0;
    g1[7] = 0;
    const i32x4 gz = {0, 0, 0, 0};
#if DMR_TDM_6ARG
    const i32x8 gz8 = {0, 0, 0, 0, 0, 0, 0, 0};
    __builtin_amdgcn_tensor_load_to_lds(g0, g1, gz, gz, gz8, 0);
#else
    __builtin_amdgcn_tensor_load_to_lds(g0, g1, gz, gz, 0);
#endif
    __builtin_amdgcn_s_wait_tensorcnt(0);
  }
#else
  // Fallback: cooperative vectorized copy with the same padded layout.
  for (int i = tid; i < 16 * (D / 4); i += 128) {
    const int r = i / (D / 4), c4 = i % (D / 4);
    const float4 v = ((const float4*)(x + (tok0 + r) * (long)D))[c4];
    *(float4*)(xs + r * ROWDW + pdw(c4 * 4)) = v;
  }
#endif
  __syncthreads();
}

// Load an A fragment (16x32 bf16, row m = lane%16) from the fp32 LDS tile.
// ISA layout: lanes 0-15 hold K = {0..7, 16..23}, lanes 16-31 K = {8..15, 24..31}.
__device__ __forceinline__ v16bf load_a_frag(const float* xs, int lmod,
                                             int half, int k0) {
  using namespace dmr;
  const float* row = xs + lmod * ROWDW;
  const int kb1 = k0 + half * 8;
  const int kb2 = k0 + 16 + half * 8;
  const float4* p1 = (const float4*)(row + pdw(kb1));  // 16B aligned
  const float4* p2 = (const float4*)(row + pdw(kb2));
  const float4 a0 = p1[0], a1 = p1[1];                 // ds_load_b128 x2
  const float4 b0 = p2[0], b1 = p2[1];                 // ds_load_b128 x2
  v16bf a;
  a[0]  = (__bf16)a0.x; a[1]  = (__bf16)a0.y; a[2]  = (__bf16)a0.z; a[3]  = (__bf16)a0.w;
  a[4]  = (__bf16)a1.x; a[5]  = (__bf16)a1.y; a[6]  = (__bf16)a1.z; a[7]  = (__bf16)a1.w;
  a[8]  = (__bf16)b0.x; a[9]  = (__bf16)b0.y; a[10] = (__bf16)b0.z; a[11] = (__bf16)b0.w;
  a[12] = (__bf16)b1.x; a[13] = (__bf16)b1.y; a[14] = (__bf16)b1.z; a[15] = (__bf16)b1.w;
  return a;
}

// B fragment: 16 strided fp32 -> bf16, column n = lmod of a [D,C]-major matrix.
__device__ __forceinline__ v16bf load_b_frag(const float* __restrict__ base,
                                             long ldc) {
  v16bf b;
#pragma unroll
  for (int i = 0; i < 16; ++i) b[i] = (__bf16)base[(long)(i < 8 ? i : i + 8) * ldc];
  return b;
}

// ---------------------------------------------------------------------------
// Kernel 1: fused predictor  score = sigmoid(relu(x@W1+b1)@W2+b2) -> dynamic_k
// grid = N/16 blocks, 128 threads (4 waves). Each wave sweeps 2 hidden chunks
// per iteration (shared A fragment -> 2 WMMAs per ds-load group).
// ---------------------------------------------------------------------------
__global__ void dmr_predictor_kernel(const float* __restrict__ x,
                                     const float* __restrict__ W1,  // [D,H]
                                     const float* __restrict__ b1,  // [H]
                                     const float* __restrict__ W2,  // [H,1]
                                     const float* __restrict__ b2,  // [1]
                                     int* __restrict__ dyn_k_out) { // [N]
  using namespace dmr;
  extern __shared__ char smem[];
  float* xs  = (float*)smem;                    // [16][ROWDW] fp32 tile
  float* ssc = (float*)(smem + XTILE_BYTES);    // [16] scores

  const int tid  = threadIdx.x;
  const int wave = tid >> 5;
  const int lane = tid & 31;
  const int lmod = lane & 15;
  const int half = lane >> 4;
  const long tok0 = (long)blockIdx.x * 16;

  if (tid < 16) ssc[tid] = 0.0f;
  stage_x_tile(x, tok0, xs, wave, tid);         // TDM DMA + barrier

  float sacc[8];
#pragma unroll
  for (int i = 0; i < 8; ++i) sacc[i] = 0.0f;

  for (int jcc = 0; jcc < H / (16 * 8); ++jcc) {  // 16 iters, 2 chunks each
    const int j0a = (jcc * 8 + wave) * 16;
    const int j0b = (jcc * 8 + wave + 4) * 16;
    v8f acc_a = {}, acc_b = {};
    for (int k0 = 0; k0 < D; k0 += 32) {
      const v16bf a = load_a_frag(xs, lmod, half, k0);
      const float* colbase = W1 + (long)(k0 + half * 8) * H;
      __builtin_prefetch(colbase + 32L * H + j0a + lmod, 0, 0);
      const v16bf ba = load_b_frag(colbase + j0a + lmod, H);
      const v16bf bb = load_b_frag(colbase + j0b + lmod, H);
      acc_a = __builtin_amdgcn_wmma_f32_16x16x32_bf16(
                  false, a, false, ba, (short)0, acc_a, false, false);
      acc_b = __builtin_amdgcn_wmma_f32_16x16x32_bf16(
                  false, a, false, bb, (short)0, acc_b, false, false);
    }
    // Fused epilogue: relu(acc + b1[j]) * W2[j], accumulated per token.
    const float w2a = W2[j0a + lmod], bba = b1[j0a + lmod];
    const float w2b = W2[j0b + lmod], bbb = b1[j0b + lmod];
#pragma unroll
    for (int i = 0; i < 8; ++i) {
      float ha = acc_a[i] + bba; ha = ha > 0.0f ? ha : 0.0f;
      float hb = acc_b[i] + bbb; hb = hb > 0.0f ? hb : 0.0f;
      sacc[i] += ha * w2a + hb * w2b;
    }
  }

  // Cross-lane / cross-wave reduction: token m = i + 8*half. (ds_add_f32)
#pragma unroll
  for (int i = 0; i < 8; ++i) atomicAdd(&ssc[i + 8 * half], sacc[i]);
  __syncthreads();

  if (tid < 16) {
    const float s   = ssc[tid] + b2[0];
    const float sig = 1.0f / (1.0f + __expf(-s));
    float k = rintf(sig * (float)TOPK) + 1.0f;          // round-half-even
    k = fminf(fmaxf(k, 1.0f), (float)TOPK);
    dyn_k_out[tok0 + tid] = (int)k;
  }
}

// ---------------------------------------------------------------------------
// Kernel 2: gating  softmax(x@W) -> top-3 -> dynamic-k mask -> outputs
// grid = N/16 blocks, 128 threads. Wave w computes experts 16w..16w+15.
// ---------------------------------------------------------------------------
__global__ void dmr_gate_kernel(const float* __restrict__ x,
                                const float* __restrict__ Wg,     // [D,E]
                                const int*   __restrict__ dyn_k,  // [N]
                                float* __restrict__ out_w,        // [N,3]
                                int*   __restrict__ out_i,        // [N,3]
                                float* __restrict__ out_m,        // [N,E]
                                int*   __restrict__ out_k) {      // [N]
  using namespace dmr;
  extern __shared__ char smem[];
  float* xs = (float*)smem;                     // [16][ROWDW] fp32 tile
  float* lg = (float*)(smem + XTILE_BYTES);     // [16][E] logits

  const int tid  = threadIdx.x;
  const int wave = tid >> 5;
  const int lane = tid & 31;
  const int lmod = lane & 15;
  const int half = lane >> 4;
  const long tok0 = (long)blockIdx.x * 16;

  stage_x_tile(x, tok0, xs, wave, tid);         // TDM DMA + barrier

  const int e0 = wave * 16;
  v8f acc = {};
  for (int k0 = 0; k0 < D; k0 += 32) {
    const v16bf a = load_a_frag(xs, lmod, half, k0);
    const v16bf b = load_b_frag(Wg + (long)(k0 + half * 8) * E + e0 + lmod, E);
    acc = __builtin_amdgcn_wmma_f32_16x16x32_bf16(
              false, a, false, b, (short)0, acc, false, false);
  }
  // Scatter logits to LDS: token m = vv + 8*half, expert e0+lmod.
#pragma unroll
  for (int vv = 0; vv < 8; ++vv)
    lg[(vv + 8 * half) * E + e0 + lmod] = acc[vv];
  __syncthreads();

  if (tid < 16) {
    const int t = tid;
    const float* row = lg + t * E;
    // softmax normalizers
    float mx = row[0];
    for (int e = 1; e < E; ++e) mx = fmaxf(mx, row[e]);
    float denom = 0.0f;
    for (int e = 0; e < E; ++e) denom += __expf(row[e] - mx);
    const float inv_denom = 1.0f / denom;
    // top-3 by logit (same order as softmax probs; ties -> lowest index)
    int   idx[TOPK];
    float prb[TOPK];
    float cut = 3.4e38f;  int cuti = -1;
    for (int s = 0; s < TOPK; ++s) {
      float best = -3.4e38f; int bi = 0;
      for (int e = 0; e < E; ++e) {
        const float v = row[e];
        const bool blocked = (v > cut) || (v == cut && e <= cuti);
        if (!blocked && v > best) { best = v; bi = e; }
      }
      idx[s] = bi;
      prb[s] = __expf(best - mx) * inv_denom;
      cut = best; cuti = bi;
    }
    // dynamic-k mask + renormalize
    const int kk = dyn_k[tok0 + t];
    float wsum = 0.0f;
#pragma unroll
    for (int s = 0; s < TOPK; ++s) { if (s >= kk) prb[s] = 0.0f; wsum += prb[s]; }
    const float inv_w = 1.0f / (wsum + 1e-8f);
    // write outputs
    float* mrow = out_m + (tok0 + t) * E;
    for (int e = 0; e < E; ++e) mrow[e] = 0.0f;
#pragma unroll
    for (int s = 0; s < TOPK; ++s) {
      const float w = prb[s] * inv_w;
      out_w[(tok0 + t) * TOPK + s] = w;
      out_i[(tok0 + t) * TOPK + s] = idx[s];
      if (prb[s] > 0.0f) mrow[idx[s]] = 1.0f;
    }
    out_k[tok0 + t] = kk;
  }
}

// ---------------------------------------------------------------------------
extern "C" void kernel_launch(void* const* d_in, const int* in_sizes, int n_in,
                              void* d_out, int out_size, void* d_ws, size_t ws_size,
                              hipStream_t stream) {
  using namespace dmr;
  (void)in_sizes; (void)n_in; (void)out_size; (void)d_ws; (void)ws_size;

  const float* x  = (const float*)d_in[0];  // [B,S,D]
  const float* Wg = (const float*)d_in[1];  // [D,E]
  const float* W1 = (const float*)d_in[2];  // [D,H]
  const float* b1 = (const float*)d_in[3];  // [H]
  const float* W2 = (const float*)d_in[4];  // [H,1]
  const float* b2 = (const float*)d_in[5];  // [1]

  // Output layout (flat, return order): weights[N,3] f32 | indices[N,3] i32 |
  // routing_mask[N,E] f32 | dynamic_top_k[N] i32
  float* out   = (float*)d_out;
  float* out_w = out;
  int*   out_i = (int*)(out + (size_t)N * TOPK);
  float* out_m = out + (size_t)N * TOPK * 2;
  int*   out_k = (int*)(out + (size_t)N * TOPK * 2 + (size_t)N * E);

  const size_t lds_pred = (size_t)XTILE_BYTES + 16 * sizeof(float);
  const size_t lds_gate = (size_t)XTILE_BYTES + 16 * E * sizeof(float);

  dim3 grid(N / 16), block(128);
  dmr_predictor_kernel<<<grid, block, lds_pred, stream>>>(x, W1, b1, W2, b2, out_k);
  dmr_gate_kernel<<<grid, block, lds_gate, stream>>>(x, Wg, out_k,
                                                     out_w, out_i, out_m, out_k);
}
